// Quantize_54640573940066
// MI455X (gfx1250) — matrix operand order repaired and checked by reading, not compiled, not run on hardware
//
#include <hip/hip_runtime.h>

// ---------------------------------------------------------------------------
// VQ-VAE quantize + EMA update for MI455X (gfx1250, wave32, WMMA).
// Distance GEMM on v_wmma_f32_16x16x32_bf16 with hi/lo bf16 splitting
// (x ~= hi + lo -> hi*hi + hi*lo + lo*hi) for ~fp32-accurate argmins.
// B fragments are shared across the 4 waves of a workgroup through LDS
// (4x less L2 traffic) and staged with async global->LDS copies
// (double-buffered, ASYNCcnt) when the toolchain exposes the builtins.
// ---------------------------------------------------------------------------

typedef __attribute__((ext_vector_type(16))) __bf16 v16bf;
typedef __attribute__((ext_vector_type(8)))  float  v8f;
typedef __attribute__((ext_vector_type(4)))  int    v4i;

#define DECAY   0.99f
#define ONE_M_D 0.01f
#define EPS_F   1e-5f

constexpr int DIM   = 256;
constexpr int NE    = 1024;       // n_embed
constexpr int NROWS = 64 * 1024;  // B*T

// ---- output layout (floats, concatenated in reference return order) ----
constexpr long Q_OFF    = 0;                         // 64*1024*256
constexpr long DIFF_OFF = (long)NROWS * DIM;         // 1
constexpr long IND_OFF  = DIFF_OFF + 1;              // NROWS (stored as float)
constexpr long NCS_OFF  = IND_OFF + NROWS;           // NE
constexpr long NEA_OFF  = NCS_OFF + NE;              // DIM*NE
constexpr long NEW_OFF  = NEA_OFF + (long)DIM * NE;  // DIM*NE

constexpr long FRAG_ELEMS = 64L * 8 * 32 * 16;       // 262144 bf16 per hi/lo

// ---- async global->LDS path availability ----
#if defined(__has_builtin)
#  if __has_builtin(__builtin_amdgcn_global_load_async_to_lds_b128) && \
      __has_builtin(__builtin_amdgcn_s_wait_asynccnt)
#    define USE_ASYNC_LDS 1
#  endif
#endif
#ifndef USE_ASYNC_LDS
#  define USE_ASYNC_LDS 0
#endif

#define AS_GLOBAL __attribute__((address_space(1)))
#define AS_LDS    __attribute__((address_space(3)))

static __device__ __forceinline__ unsigned short f32_to_bf16_rne(float f) {
    unsigned int u = __float_as_uint(f);
    unsigned int r = (u + 0x7FFFu + ((u >> 16) & 1u)) >> 16;
    return (unsigned short)r;
}
static __device__ __forceinline__ float bf16u_to_f32(unsigned short h) {
    return __uint_as_float((unsigned int)h << 16);
}

union Frag16 { v16bf v; unsigned short u[16]; };
union FragQ  { uint4 q[2]; v16bf v; };

// ---------------------------------------------------------------------------
// Kernel Z: zero scratch accumulators (counts, esum, diffacc, nval)
// ---------------------------------------------------------------------------
__global__ void vq_zero_ws(float* p, long n) {
    long i = (long)blockIdx.x * blockDim.x + threadIdx.x;
    if (i < n) p[i] = 0.0f;
}

// ---------------------------------------------------------------------------
// Kernel A1: split embed[256x1024] f32 into bf16 hi/lo, pre-swizzled into
// WMMA B-fragment order: frag idx = ((ct*8+kb)*32 + lane)*16 + j
//   N = ct*16 + (lane&15), K = kb*32 + (j>>3)*16 + (lane>>4)*8 + (j&7)
// ---------------------------------------------------------------------------
__global__ void vq_make_bfrags(const float* __restrict__ embed,
                               unsigned short* __restrict__ e_hi,
                               unsigned short* __restrict__ e_lo) {
    long t = (long)blockIdx.x * blockDim.x + threadIdx.x;
    if (t >= FRAG_ELEMS) return;
    int j    = (int)(t & 15);
    int lane = (int)((t >> 4) & 31);
    int kb   = (int)((t >> 9) & 7);
    int ct   = (int)(t >> 12);
    int K = kb * 32 + (j >> 3) * 16 + (lane >> 4) * 8 + (j & 7);
    int N = ct * 16 + (lane & 15);
    float val = embed[(long)K * NE + N];
    unsigned short hi = f32_to_bf16_rne(val);
    float lo = val - bf16u_to_f32(hi);
    e_hi[t] = hi;
    e_lo[t] = f32_to_bf16_rne(lo);
}

// ---------------------------------------------------------------------------
// Kernel A2: enorm[c] = sum_d embed[d,c]^2
// ---------------------------------------------------------------------------
__global__ void vq_enorm(const float* __restrict__ embed, float* __restrict__ enorm) {
    int c = blockIdx.x * blockDim.x + threadIdx.x;
    if (c >= NE) return;
    float s = 0.0f;
    for (int d = 0; d < DIM; ++d) {
        float e = embed[(long)d * NE + c];
        s += e * e;
    }
    enorm[c] = s;
}

// ---------------------------------------------------------------------------
// Kernel B: fused distance-GEMM + argmin + quantize gather + diff + scatter.
// 128 threads = 4 waves; each wave owns 16 rows; B tiles shared via LDS with
// double-buffered async global->LDS staging.
// ---------------------------------------------------------------------------
#define XPITCH 257   // LDS row pitch (floats) -> conflict-free column reads

__global__ __launch_bounds__(128) void vq_main(
    const float* __restrict__ x,            // [NROWS, DIM]
    const float* __restrict__ embed,        // [DIM, NE]
    const unsigned short* __restrict__ e_hi,
    const unsigned short* __restrict__ e_lo,
    const float* __restrict__ enorm,
    float* __restrict__ counts,
    float* __restrict__ esum,               // [DIM, NE]
    float* __restrict__ diffacc,
    float* __restrict__ out)                // full output buffer
{
    __shared__ __align__(16) float lds_x[4 * 16 * XPITCH];      // 64.25 KB
    __shared__ __align__(16) unsigned char lds_b[2][16384];     // 32 KB (hi|lo)
    __shared__ float lds_xn[4 * 16];
    __shared__ int   lds_ind[4 * 16];

    const int wave = threadIdx.x >> 5;
    const int lane = threadIdx.x & 31;
    const int m    = lane & 15;      // row-in-tile for A / N-lane for B,C
    const int lg   = lane >> 4;      // half-wave group
    float* xt = lds_x + wave * 16 * XPITCH;
    const long rowbase = ((long)blockIdx.x * 4 + wave) * 16;

    const unsigned char* ehi8 = (const unsigned char*)e_hi;
    const unsigned char* elo8 = (const unsigned char*)e_lo;

    // ---- B-tile staging: 16 KB per ct (8 KB hi + 8 KB lo), 128 threads ----
    // chunk c (16B): which=c>>9; rem=c&511; kb=rem>>6; ln=(rem&63)>>1; half=rem&1
    // global byte  : ct*8192 + kb*1024 + ln*32 + half*16     (fully coalesced)
    // lds byte     : which*8192 + (kb*2+half)*512 + ln*16
    auto stageB = [&](int ct, int buf) {
        #pragma unroll
        for (int it = 0; it < 8; ++it) {
            int c     = it * 128 + (int)threadIdx.x;   // 0..1023
            int which = c >> 9;
            int rem   = c & 511;
            int kb    = rem >> 6;
            int r2    = rem & 63;
            int ln    = r2 >> 1;
            int half  = r2 & 1;
            long goff = (long)ct * 8192 + kb * 1024 + ln * 32 + half * 16;
            int  loff = which * 8192 + (kb * 2 + half) * 512 + ln * 16;
            const unsigned char* gsrc = (which ? elo8 : ehi8) + goff;
            unsigned char* dst = &lds_b[buf][loff];
#if USE_ASYNC_LDS
            __builtin_amdgcn_global_load_async_to_lds_b128(
                (AS_GLOBAL v4i*)gsrc, (AS_LDS v4i*)dst, 0, 0);
#else
            *(uint4*)dst = *(const uint4*)gsrc;
#endif
        }
    };

    // ---- stage 16x256 f32 rows into LDS (coalesced float4) ----
    const float4* src = (const float4*)(x + rowbase * DIM);
    for (int it = 0; it < 32; ++it) {
        int flat = it * 32 + lane;          // float4 index, 0..1023
        int r  = flat >> 6;                 // 64 float4 per row
        int c4 = flat & 63;
        float4 v = src[r * 64 + c4];
        float* dst = xt + r * XPITCH + c4 * 4;
        dst[0] = v.x; dst[1] = v.y; dst[2] = v.z; dst[3] = v.w;
    }

    // ---- kick off first B tile, then make x tile + B tile visible ----
    stageB(0, 0);
#if USE_ASYNC_LDS
    __builtin_amdgcn_s_wait_asynccnt(0);
#endif
    __syncthreads();

    // ---- per-row squared norms (lanes 0..15 each do one row) ----
    if (lane < 16) {
        const float* rp = xt + lane * XPITCH;
        float s = 0.0f;
        for (int d = 0; d < DIM; ++d) s += rp[d] * rp[d];
        lds_xn[wave * 16 + lane] = s;
    }
    __syncthreads();
    float xnorm8[8];
    for (int i = 0; i < 8; ++i) xnorm8[i] = lds_xn[wave * 16 + lg * 8 + i];

    // ---- build A fragments (ISA 16-bit A 16x32 layout), hi/lo split ----
    v16bf a_hi[8], a_lo[8];
    for (int kb = 0; kb < 8; ++kb) {
        Frag16 fh, fl;
        for (int j = 0; j < 16; ++j) {
            int v = j >> 1, h = j & 1;
            int K = kb * 32 + ((v >= 4) ? 16 : 0) + (v & 3) * 2 + h + lg * 8;
            float val = xt[m * XPITCH + K];
            unsigned short hi = f32_to_bf16_rne(val);
            fh.u[j] = hi;
            fl.u[j] = f32_to_bf16_rne(val - bf16u_to_f32(hi));
        }
        a_hi[kb] = fh.v;
        a_lo[kb] = fl.v;
    }

    // ---- distance GEMM + running argmin over 64 column tiles ----
    float minv[8];
    int   mini[8];
    for (int i = 0; i < 8; ++i) { minv[i] = 3.4e38f; mini[i] = 0; }

    for (int ct = 0; ct < 64; ++ct) {
        int cur = ct & 1;
        if (ct + 1 < 64) stageB(ct + 1, cur ^ 1);   // overlap next tile copy

        v8f acc = {};
        #pragma unroll
        for (int kb = 0; kb < 8; ++kb) {
            FragQ bh, bl;
            const unsigned char* bb = lds_b[cur];
            bh.q[0] = *(const uint4*)(bb + (kb * 2 + 0) * 512 + lane * 16);
            bh.q[1] = *(const uint4*)(bb + (kb * 2 + 1) * 512 + lane * 16);
            bl.q[0] = *(const uint4*)(bb + 8192 + (kb * 2 + 0) * 512 + lane * 16);
            bl.q[1] = *(const uint4*)(bb + 8192 + (kb * 2 + 1) * 512 + lane * 16);
            acc = __builtin_amdgcn_wmma_f32_16x16x32_bf16(
                false, a_hi[kb], false, bh.v, (short)0, acc, false, false);
            acc = __builtin_amdgcn_wmma_f32_16x16x32_bf16(
                false, a_hi[kb], false, bl.v, (short)0, acc, false, false);
            acc = __builtin_amdgcn_wmma_f32_16x16x32_bf16(
                false, a_lo[kb], false, bh.v, (short)0, acc, false, false);
        }
        int col = ct * 16 + m;
        float en = enorm[col];
        #pragma unroll
        for (int i = 0; i < 8; ++i) {
            float dist = xnorm8[i] - 2.0f * acc[i] + en;
            if (dist < minv[i]) { minv[i] = dist; mini[i] = col; }
        }

#if USE_ASYNC_LDS
        __builtin_amdgcn_s_wait_asynccnt(0);
#endif
        __syncthreads();   // next buffer ready for all; this buffer reusable
    }

    // ---- reduce (min, argmin) across the 16 lanes of each half-wave ----
    for (int off = 1; off < 16; off <<= 1) {
        #pragma unroll
        for (int i = 0; i < 8; ++i) {
            float ov = __shfl_xor(minv[i], off, 16);
            int   oi = __shfl_xor(mini[i], off, 16);
            if (ov < minv[i] || (ov == minv[i] && oi < mini[i])) {
                minv[i] = ov; mini[i] = oi;
            }
        }
    }
    if (m == 0) {   // lanes 0 and 16 hold rows lg*8 + i
        for (int i = 0; i < 8; ++i) {
            int r = lg * 8 + i;
            int idx = mini[i];
            lds_ind[wave * 16 + r] = idx;
            out[IND_OFF + rowbase + r] = (float)idx;
            atomicAdd(&counts[idx], 1.0f);
        }
    }
    __syncthreads();

    // ---- quantize gather + diff + segment-sum scatter (reuse LDS x) ----
    float dacc = 0.0f;
    for (int it = 0; it < 128; ++it) {
        int r = it >> 3;
        int d = ((it & 7) << 5) + lane;
        int idx = lds_ind[wave * 16 + r];
        float q  = embed[(long)d * NE + idx];
        float xv = xt[r * XPITCH + d];
        out[Q_OFF + (rowbase + r) * DIM + d] = q;
        float dd = q - xv;
        dacc += dd * dd;
        atomicAdd(&esum[(long)d * NE + idx], xv);
    }
    for (int off = 1; off < 32; off <<= 1) dacc += __shfl_xor(dacc, off, 32);
    if (lane == 0) atomicAdd(diffacc, dacc);
}

// ---------------------------------------------------------------------------
// Kernel C1: new_cluster_size + n reduction + diff scalar. One 1024-thr block.
// ---------------------------------------------------------------------------
__global__ void vq_ema_cs(const float* __restrict__ cluster_size,
                          const float* __restrict__ counts,
                          const float* __restrict__ diffacc,
                          float* __restrict__ nval,
                          float* __restrict__ out) {
    __shared__ float red[1024];
    int c = threadIdx.x;
    float ncs = cluster_size[c] * DECAY + ONE_M_D * counts[c];
    out[NCS_OFF + c] = ncs;
    red[c] = ncs;
    __syncthreads();
    for (int s = 512; s > 0; s >>= 1) {
        if (c < s) red[c] += red[c + s];
        __syncthreads();
    }
    if (c == 0) {
        nval[0] = red[0];
        out[DIFF_OFF] = diffacc[0] / (float)((long)NROWS * DIM);
    }
}

// ---------------------------------------------------------------------------
// Kernel C2: new_embed_avg (decay^2, faithful to reference) and new_embed.
// ---------------------------------------------------------------------------
__global__ void vq_ema_embed(const float* __restrict__ embed_avg,
                             const float* __restrict__ esum,
                             const float* __restrict__ nval,
                             float* __restrict__ out) {
    long i = (long)blockIdx.x * blockDim.x + threadIdx.x;
    if (i >= (long)DIM * NE) return;
    int c = (int)(i & (NE - 1));
    float nea = embed_avg[i] * (DECAY * DECAY) + ONE_M_D * esum[i];
    out[NEA_OFF + i] = nea;
    float n = nval[0];
    float ncs = out[NCS_OFF + c];
    float cs = (ncs + EPS_F) / (n + (float)NE * EPS_F) * n;
    out[NEW_OFF + i] = nea / cs;
}

// ---------------------------------------------------------------------------
extern "C" void kernel_launch(void* const* d_in, const int* in_sizes, int n_in,
                              void* d_out, int out_size, void* d_ws, size_t ws_size,
                              hipStream_t stream) {
    const float* input_       = (const float*)d_in[0]; // [64,1024,256]
    const float* embed        = (const float*)d_in[1]; // [256,1024]
    const float* cluster_size = (const float*)d_in[2]; // [1024]
    const float* embed_avg    = (const float*)d_in[3]; // [256,1024]
    float* out = (float*)d_out;

    // workspace carve-up (~2.1 MB)
    unsigned short* e_hi = (unsigned short*)d_ws;                // 262144 bf16
    unsigned short* e_lo = e_hi + FRAG_ELEMS;                    // 262144 bf16
    float* enorm   = (float*)(e_lo + FRAG_ELEMS);                // 1024
    float* counts  = enorm + NE;                                 // 1024
    float* esum    = counts + NE;                                // 262144
    float* diffacc = esum + (long)DIM * NE;                      // 1
    float* nval    = diffacc + 1;                                // 1

    // Z: zero accumulators (counts + esum + diffacc + nval)
    long nz = NE + (long)DIM * NE + 2;
    vq_zero_ws<<<(unsigned)((nz + 255) / 256), 256, 0, stream>>>(counts, nz);

    // A: prebuild bf16 hi/lo B-fragments and column norms
    vq_make_bfrags<<<(unsigned)(FRAG_ELEMS / 256), 256, 0, stream>>>(embed, e_hi, e_lo);
    vq_enorm<<<NE / 256, 256, 0, stream>>>(embed, enorm);

    // B: main fused kernel — 4 waves/block, 16 rows/wave, LDS-shared B tiles
    vq_main<<<NROWS / 64, 128, 0, stream>>>(input_, embed, e_hi, e_lo, enorm,
                                            counts, esum, diffacc, out);

    // C: EMA finalize
    vq_ema_cs<<<1, 1024, 0, stream>>>(cluster_size, counts, diffacc, nval, out);
    vq_ema_embed<<<(unsigned)(((long)DIM * NE + 255) / 256), 256, 0, stream>>>(
        embed_avg, esum, nval, out);
}
